// MolecularGCN_87514253623366
// MI455X (gfx1250) — compile-verified, matrix operand order checked by reading, not run on hardware
//
#include <hip/hip_runtime.h>
#include <hip/hip_bf16.h>

#define NNODES   100000
#define NEDGES   3200000
#define NGRAPHS  1024
#define NODE_DIM 32
#define EDGE_DIM 16
#define HID      128
#define NLAYERS  3

typedef __bf16 bf16;
typedef __attribute__((ext_vector_type(16))) bf16  v16bf;
typedef __attribute__((ext_vector_type(8)))  bf16  v8bf;
typedef __attribute__((ext_vector_type(4)))  bf16  v4bf;
typedef __attribute__((ext_vector_type(8)))  float v8f;
typedef __attribute__((ext_vector_type(8)))  float v8fl;
typedef __attribute__((ext_vector_type(4)))  float v4fl;

// ---------- bf16 helpers (native hardware converts) ----------
__device__ __forceinline__ bf16  f2bf(float f) { return (bf16)f; }
__device__ __forceinline__ float bf2f(bf16 b)  { return (float)b; }

__device__ __forceinline__ v8f wmma_bf16(v16bf a, v16bf b, v8f c) {
  return __builtin_amdgcn_wmma_f32_16x16x32_bf16(false, a, false, b, (short)0, c, false, false);
}

// Build v16bf A-register from two contiguous 8-float runs (k0..k0+7, k0+16..k0+23)
__device__ __forceinline__ v16bf cvt8x2(v8fl lo, v8fl hi) {
  v16bf a;
#pragma unroll
  for (int i = 0; i < 8; ++i) { a[i] = f2bf(lo[i]); a[8 + i] = f2bf(hi[i]); }
  return a;
}
__device__ __forceinline__ v16bf join8x2(v8bf lo, v8bf hi) {
  v16bf a;
#pragma unroll
  for (int i = 0; i < 8; ++i) { a[i] = lo[i]; a[8 + i] = hi[i]; }
  return a;
}

// Packed B tile: one contiguous 32B vector load per lane (wave reads 1KB contiguous)
__device__ __forceinline__ v16bf load_b_packed(const bf16* __restrict__ pk, int tile, int lane) {
  return *(const v16bf*)(pk + (((size_t)tile * 32 + lane) << 4));
}

// ---------- weight repack: fp32 [K,N] -> bf16 packed B-tile layout, K zero-padded to Kp ----------
__global__ void k_pack_b(const float* __restrict__ src, bf16* __restrict__ dst,
                         int K, int Kp, int N) {
  int o = blockIdx.x * blockDim.x + threadIdx.x;
  int total = Kp * N;
  if (o >= total) return;
  int tile = o >> 9;            // 512 elements per 32x16 tile
  int lane = (o >> 4) & 31;
  int e    = o & 15;
  int tiles_k = Kp >> 5;
  int nt = tile / tiles_k, c = tile % tiles_k;
  int hlf = lane >> 4, nl = lane & 15;
  int k = c * 32 + ((e & 8) << 1) + hlf * 8 + (e & 7);
  int n = nt * 16 + nl;
  float v = (k < K) ? src[(size_t)k * N + n] : 0.0f;
  dst[o] = f2bf(v);
}

__global__ void k_zero4(float* __restrict__ p, long long n4) {
  long long i = (long long)blockIdx.x * blockDim.x + threadIdx.x;
  if (i < n4) *(v4fl*)(p + i * 4) = v4fl{0.0f, 0.0f, 0.0f, 0.0f};
}

// ---------- node encoder: x = relu(nf @ Wn + bn), K=32 ----------
__global__ void k_node_encoder(const float* __restrict__ nf, const bf16* __restrict__ wn_pk,
                               const float* __restrict__ bias, float* __restrict__ x) {
  int wid  = (blockIdx.x * blockDim.x + threadIdx.x) >> 5;
  int lane = threadIdx.x & 31;
  if (wid >= NNODES / 16) return;               // wave-uniform
  int m0 = wid * 16;
  int m = lane & 15, hlf = lane >> 4, n = lane & 15;
  const float* ap = nf + (size_t)(m0 + m) * NODE_DIM + hlf * 8;
  v16bf a = cvt8x2(__builtin_nontemporal_load((const v8fl*)ap),
                   __builtin_nontemporal_load((const v8fl*)(ap + 16)));
#pragma unroll
  for (int nt = 0; nt < 8; ++nt) {
    v16bf b = load_b_packed(wn_pk, nt, lane);
    v8f acc = {};
    acc = wmma_bf16(a, b, acc);
    float bv = bias[nt * 16 + n];
#pragma unroll
    for (int r = 0; r < 8; ++r) {
      int mm = r + 8 * hlf;
      x[(size_t)(m0 + mm) * HID + nt * 16 + n] = fmaxf(acc[r] + bv, 0.0f);
    }
  }
}

// ---------- edge encoder: e = ef @ We + be (no relu), K=16 padded to 32, out bf16 NT ----------
__global__ void k_edge_encoder(const float* __restrict__ ef, const bf16* __restrict__ we_pk,
                               const float* __restrict__ bias, bf16* __restrict__ e) {
  int wid  = (blockIdx.x * blockDim.x + threadIdx.x) >> 5;
  int lane = threadIdx.x & 31;
  if (wid >= NEDGES / 16) return;               // wave-uniform
  int m0 = wid * 16;
  int m = lane & 15, hlf = lane >> 4, n = lane & 15;
  const float* ap = ef + (size_t)(m0 + m) * EDGE_DIM + hlf * 8;   // k in [0,16)
  v8fl hz = {};
  v16bf a = cvt8x2(__builtin_nontemporal_load((const v8fl*)ap), hz);  // upper-K zeros
#pragma unroll
  for (int nt = 0; nt < 8; ++nt) {
    v16bf b = load_b_packed(we_pk, nt, lane);   // zero-padded at pack time
    v8f acc = {};
    acc = wmma_bf16(a, b, acc);
    float bv = bias[nt * 16 + n];
#pragma unroll
    for (int r = 0; r < 8; ++r) {
      int mm = r + 8 * hlf;
      __builtin_nontemporal_store(f2bf(acc[r] + bv),
                                  e + (size_t)(m0 + mm) * HID + nt * 16 + n);
    }
  }
}

// ---------- fused message pass: agg[dst] += relu(x[src] + e), e streamed NT ----------
__global__ void k_message(const float* __restrict__ x, const bf16* __restrict__ e,
                          const int* __restrict__ src, const int* __restrict__ dst,
                          float* __restrict__ agg) {
  long long t = (long long)blockIdx.x * blockDim.x + threadIdx.x;
  if (t >= (long long)NEDGES * 32) return;
  int eIdx = (int)(t >> 5);
  int f    = ((int)t & 31) * 4;
  int s = src[eIdx], d = dst[eIdx];
  v4fl xs = *(const v4fl*)(x + (size_t)s * HID + f);                 // keep cached (L2-resident)
  v4bf ev = __builtin_nontemporal_load((const v4bf*)(e + (size_t)eIdx * HID + f));
  float* ap = agg + (size_t)d * HID + f;
#pragma unroll
  for (int i = 0; i < 4; ++i) {
    float m = fmaxf(xs[i] + bf2f(ev[i]), 0.0f);
    atomicAdd(ap + i, m);
  }
}

// ---------- fused GINE MLP: x = relu(relu(((1+eps)x+agg)@W1+b1)@W2+b2) ----------
__global__ __launch_bounds__(128) void k_layer_mlp(
    float* __restrict__ x, const float* __restrict__ agg,
    const bf16* __restrict__ w1_pk, const float* __restrict__ b1,
    const bf16* __restrict__ w2_pk, const float* __restrict__ b2,
    const float* __restrict__ eps, int layer) {
  __shared__ bf16 lds[4][16 * HID];             // 4KB per wave
  int wid  = (blockIdx.x * blockDim.x + threadIdx.x) >> 5;
  int lane = threadIdx.x & 31;
  if (wid >= NNODES / 16) return;               // wave-uniform
  bf16* ldsw = lds[(threadIdx.x >> 5) & 3];
  int m0 = wid * 16;
  int m = lane & 15, hlf = lane >> 4, n = lane & 15;
  float e1 = 1.0f + eps[layer];

  // stage combine (1+eps)*x + agg into LDS bf16, coalesced float4 global loads
#pragma unroll
  for (int j = 0; j < 16; ++j) {
    size_t base = (size_t)(m0 + j) * HID + lane * 4;
    v4fl xv = *(const v4fl*)(x + base);
    v4fl av = *(const v4fl*)(agg + base);
    v4bf tv;
#pragma unroll
    for (int i = 0; i < 4; ++i) tv[i] = f2bf(e1 * xv[i] + av[i]);
    *(v4bf*)(ldsw + j * HID + lane * 4) = tv;   // ds_store_b64
  }

  // A from LDS via contiguous ds_load_b128 pairs (same-wave DS is in-order)
  v16bf a[4];
#pragma unroll
  for (int c = 0; c < 4; ++c) {
    const bf16* p = ldsw + m * HID + c * 32 + hlf * 8;
    a[c] = join8x2(*(const v8bf*)p, *(const v8bf*)(p + 16));
  }

  // GEMM1 + bias + relu -> LDS (reuse region; DS in-order keeps RAW/WAR safe)
#pragma unroll
  for (int nt = 0; nt < 8; ++nt) {
    v8f acc = {};
#pragma unroll
    for (int c = 0; c < 4; ++c)
      acc = wmma_bf16(a[c], load_b_packed(w1_pk, nt * 4 + c, lane), acc);
    float bv = b1[nt * 16 + n];
#pragma unroll
    for (int r = 0; r < 8; ++r) {
      int mm = r + 8 * hlf;
      ldsw[mm * HID + nt * 16 + n] = f2bf(fmaxf(acc[r] + bv, 0.0f));
    }
  }

  // hidden back in A-layout
  v16bf a2[4];
#pragma unroll
  for (int c = 0; c < 4; ++c) {
    const bf16* p = ldsw + m * HID + c * 32 + hlf * 8;
    a2[c] = join8x2(*(const v8bf*)p, *(const v8bf*)(p + 16));
  }

  // GEMM2 + bias + relu -> x (in place; this wave owns rows m0..m0+15)
#pragma unroll
  for (int nt = 0; nt < 8; ++nt) {
    v8f acc = {};
#pragma unroll
    for (int c = 0; c < 4; ++c)
      acc = wmma_bf16(a2[c], load_b_packed(w2_pk, nt * 4 + c, lane), acc);
    float bv = b2[nt * 16 + n];
#pragma unroll
    for (int r = 0; r < 8; ++r) {
      int mm = r + 8 * hlf;
      x[(size_t)(m0 + mm) * HID + nt * 16 + n] = fmaxf(acc[r] + bv, 0.0f);
    }
  }
}

// ---------- global mean pool (sum + counts via atomics) ----------
__global__ void k_pool(const float* __restrict__ x, const int* __restrict__ batch,
                       float* __restrict__ pooled, float* __restrict__ counts) {
  long long t = (long long)blockIdx.x * blockDim.x + threadIdx.x;
  if (t >= (long long)NNODES * 32) return;
  int node = (int)(t >> 5);
  int f    = ((int)t & 31) * 4;
  int b = batch[node];
  v4fl xv = *(const v4fl*)(x + (size_t)node * HID + f);
  float* pp = pooled + (size_t)b * HID + f;
#pragma unroll
  for (int i = 0; i < 4; ++i) atomicAdd(pp + i, xv[i]);
  if (f == 0) atomicAdd(counts + b, 1.0f);
}

// ---------- readout GEMM: h = relu((pooled/cnt) @ fc_w1 + fc_b1) ----------
__global__ void k_readout1(const float* __restrict__ pooled, const float* __restrict__ counts,
                           const bf16* __restrict__ fc1_pk, const float* __restrict__ fb1,
                           float* __restrict__ h) {
  int wid  = (blockIdx.x * blockDim.x + threadIdx.x) >> 5;
  int lane = threadIdx.x & 31;
  if (wid >= NGRAPHS / 16) return;              // wave-uniform
  int m0 = wid * 16;
  int m = lane & 15, hlf = lane >> 4, n = lane & 15;
  int row = m0 + m;
  float inv = 1.0f / fmaxf(counts[row], 1.0f);
  v16bf a[4];
#pragma unroll
  for (int c = 0; c < 4; ++c) {
    const float* p = pooled + (size_t)row * HID + c * 32 + hlf * 8;
    v8fl lo = *(const v8fl*)p, hi = *(const v8fl*)(p + 16);
#pragma unroll
    for (int i = 0; i < 8; ++i) { lo[i] *= inv; hi[i] *= inv; }
    a[c] = cvt8x2(lo, hi);
  }
#pragma unroll
  for (int nt = 0; nt < 8; ++nt) {
    v8f acc = {};
#pragma unroll
    for (int c = 0; c < 4; ++c)
      acc = wmma_bf16(a[c], load_b_packed(fc1_pk, nt * 4 + c, lane), acc);
    float bv = fb1[nt * 16 + n];
#pragma unroll
    for (int r = 0; r < 8; ++r) {
      int mm = r + 8 * hlf;
      h[(size_t)(m0 + mm) * HID + nt * 16 + n] = fmaxf(acc[r] + bv, 0.0f);
    }
  }
}

// ---------- readout matvec: out = h @ fc_w2 + fc_b2 ----------
__global__ void k_readout2(const float* __restrict__ h, const float* __restrict__ fw2,
                           const float* __restrict__ fb2, float* __restrict__ out) {
  int wid  = (blockIdx.x * blockDim.x + threadIdx.x) >> 5;
  int lane = threadIdx.x & 31;
  if (wid >= NGRAPHS) return;
  float s = 0.0f;
#pragma unroll
  for (int k = lane; k < HID; k += 32) s += h[(size_t)wid * HID + k] * fw2[k];
#pragma unroll
  for (int o = 16; o > 0; o >>= 1) s += __shfl_down(s, o, 32);
  if (lane == 0) out[wid] = s + fb2[0];
}

extern "C" void kernel_launch(void* const* d_in, const int* in_sizes, int n_in,
                              void* d_out, int out_size, void* d_ws, size_t ws_size,
                              hipStream_t stream) {
  (void)in_sizes; (void)n_in; (void)out_size; (void)ws_size;
  const float* nf    = (const float*)d_in[0];
  const float* ef    = (const float*)d_in[1];
  const int*   eidx  = (const int*)d_in[2];
  const int*   batch = (const int*)d_in[3];
  const float* Wn  = (const float*)d_in[4];
  const float* bn  = (const float*)d_in[5];
  const float* We  = (const float*)d_in[6];
  const float* be  = (const float*)d_in[7];
  const float* w1  = (const float*)d_in[8];
  const float* b1  = (const float*)d_in[9];
  const float* w2  = (const float*)d_in[10];
  const float* b2  = (const float*)d_in[11];
  const float* eps = (const float*)d_in[12];
  const float* fw1 = (const float*)d_in[13];
  const float* fb1 = (const float*)d_in[14];
  const float* fw2 = (const float*)d_in[15];
  const float* fb2 = (const float*)d_in[16];
  float* out = (float*)d_out;

  // workspace carve-up
  char* ws = (char*)d_ws;
  size_t off = 0;
  auto take = [&](size_t bytes) { size_t o = off; off = (off + bytes + 255) & ~(size_t)255; return o; };
  bf16*  e_bf   = (bf16*)(ws + take((size_t)NEDGES * HID * 2));
  float* x      = (float*)(ws + take((size_t)NNODES * HID * 4));
  float* agg    = (float*)(ws + take((size_t)NNODES * HID * 4));
  bf16*  wn_pk  = (bf16*)(ws + take((size_t)32 * HID * 2));
  bf16*  we_pk  = (bf16*)(ws + take((size_t)32 * HID * 2));          // K padded 16->32
  bf16*  w1_pk  = (bf16*)(ws + take((size_t)NLAYERS * HID * HID * 2));
  bf16*  w2_pk  = (bf16*)(ws + take((size_t)NLAYERS * HID * HID * 2));
  bf16*  fc1_pk = (bf16*)(ws + take((size_t)HID * HID * 2));
  float* pooled = (float*)(ws + take((size_t)NGRAPHS * HID * 4));
  float* counts = (float*)(ws + take((size_t)NGRAPHS * 4));
  float* hred   = (float*)(ws + take((size_t)NGRAPHS * HID * 4));

  const int* src = eidx;
  const int* dst = eidx + NEDGES;

  // 1) repack weights into WMMA B-tile layout (bf16)
  k_pack_b<<<(32 * HID + 255) / 256, 256, 0, stream>>>(Wn, wn_pk, 32, 32, HID);
  k_pack_b<<<(32 * HID + 255) / 256, 256, 0, stream>>>(We, we_pk, 16, 32, HID);
  for (int l = 0; l < NLAYERS; ++l) {
    k_pack_b<<<(HID * HID + 255) / 256, 256, 0, stream>>>(
        w1 + (size_t)l * HID * HID, w1_pk + (size_t)l * HID * HID, HID, HID, HID);
    k_pack_b<<<(HID * HID + 255) / 256, 256, 0, stream>>>(
        w2 + (size_t)l * HID * HID, w2_pk + (size_t)l * HID * HID, HID, HID, HID);
  }
  k_pack_b<<<(HID * HID + 255) / 256, 256, 0, stream>>>(fw1, fc1_pk, HID, HID, HID);

  // 2) encoders
  k_node_encoder<<<((NNODES / 16) * 32 + 255) / 256, 256, 0, stream>>>(nf, wn_pk, bn, x);
  k_edge_encoder<<<((NEDGES / 16) * 32 + 255) / 256, 256, 0, stream>>>(ef, we_pk, be, e_bf);

  // 3) GINE layers
  for (int l = 0; l < NLAYERS; ++l) {
    long long nagg4 = (long long)NNODES * HID / 4;
    k_zero4<<<(unsigned)((nagg4 + 255) / 256), 256, 0, stream>>>(agg, nagg4);
    long long nmsg = (long long)NEDGES * 32;
    k_message<<<(unsigned)((nmsg + 255) / 256), 256, 0, stream>>>(x, e_bf, src, dst, agg);
    k_layer_mlp<<<(NNODES / 16 + 3) / 4, 128, 0, stream>>>(
        x, agg, w1_pk + (size_t)l * HID * HID, b1 + (size_t)l * HID,
        w2_pk + (size_t)l * HID * HID, b2 + (size_t)l * HID, eps, l);
  }

  // 4) pool
  k_zero4<<<((NGRAPHS * HID / 4) + 255) / 256, 256, 0, stream>>>(pooled, (long long)NGRAPHS * HID / 4);
  k_zero4<<<((NGRAPHS / 4) + 255) / 256, 256, 0, stream>>>(counts, (long long)NGRAPHS / 4);
  long long npool = (long long)NNODES * 32;
  k_pool<<<(unsigned)((npool + 255) / 256), 256, 0, stream>>>(x, batch, pooled, counts);

  // 5) readout
  k_readout1<<<((NGRAPHS / 16) * 32 + 255) / 256, 256, 0, stream>>>(pooled, counts, fc1_pk, fb1, hred);
  k_readout2<<<(NGRAPHS * 32 + 255) / 256, 256, 0, stream>>>(hred, fw2, fb2, out);
}
